// BCosGCN_28346784153649
// MI455X (gfx1250) — matrix operand order repaired and checked by reading, compiled, not run on hardware
//
#include <hip/hip_runtime.h>
#include <hip/hip_bf16.h>

typedef __attribute__((ext_vector_type(16))) __bf16 v16bf;
typedef __attribute__((ext_vector_type(8)))  float  v8f;

#define LN_EPS   1e-5f
#define BCOS_EPS 1e-6f
#define TEMP_C   1.5f
#define RR_C     0.6f

__device__ __forceinline__ unsigned short f2bf(float f) {
  unsigned int u = __float_as_uint(f);
  unsigned int r = u + 0x7FFFu + ((u >> 16) & 1u);   // round-to-nearest-even
  return (unsigned short)(r >> 16);
}

__device__ __forceinline__ float wave_sum32(float v) {
#pragma unroll
  for (int off = 16; off > 0; off >>= 1) v += __shfl_xor(v, off, 32);
  return v;
}

// ---------------------------------------------------------------- utility
// n must be a multiple of 4
__global__ void zero_f32v4(float4* __restrict__ p, long long n4) {
  long long i = (long long)blockIdx.x * blockDim.x + threadIdx.x;
  long long stride = (long long)gridDim.x * blockDim.x;
  float4 z = {0.f, 0.f, 0.f, 0.f};
  for (; i < n4; i += stride) p[i] = z;
}

// n must be a multiple of 4
__global__ void cvt_f32_to_bf16v4(const float4* __restrict__ in,
                                  ushort4* __restrict__ out, long long n4) {
  long long i = (long long)blockIdx.x * blockDim.x + threadIdx.x;
  long long stride = (long long)gridDim.x * blockDim.x;
  for (; i < n4; i += stride) {
    float4 v = in[i];
    ushort4 o;
    o.x = f2bf(v.x); o.y = f2bf(v.y); o.z = f2bf(v.z); o.w = f2bf(v.w);
    out[i] = o;
  }
}

// W [rows, cols] f32 row-major -> Wt [cols, rows] bf16 row-major
__global__ void transpose_w_bf16(const float* __restrict__ W,
                                 unsigned short* __restrict__ Wt,
                                 int rows, int cols) {
  int i = blockIdx.x * blockDim.x + threadIdx.x;
  int n = rows * cols;
  if (i >= n) return;
  int f = i / cols, h = i % cols;
  Wt[h * rows + f] = f2bf(W[i]);
}

// ---------------------------------------------------------------- degree
__global__ void edge_degree(const int* __restrict__ dst, float* __restrict__ deg, int E) {
  int e = blockIdx.x * blockDim.x + threadIdx.x;
  if (e < E) atomicAdd(deg + dst[e], 1.0f);
}

__global__ void degree_to_dinv(float* __restrict__ d, int N) {
  int i = blockIdx.x * blockDim.x + threadIdx.x;
  if (i < N) d[i] = rsqrtf(d[i] + 1.0f);   // self-loop adds 1; deg >= 1
}

// ---------------------------------------------------------------- WMMA GEMM
// out[M,H] = A[M,K](bf16) @ Bt[H,K](bf16)^T, f32 accumulate. All-whole-tile
// fast path: each wave computes MT stacked 16x16 tiles sharing one B fragment.
// grid.x = row blocks of 16*MT; block = 256 (8 waves); wave w -> cols [16w,16w+16)
template <int K, int MT>
__global__ void gemm_bf16_wmma(const unsigned short* __restrict__ A,
                               const unsigned short* __restrict__ Bt,
                               float* __restrict__ out, int Hcols) {
  const int lane = threadIdx.x & 31;
  const int wave = threadIdx.x >> 5;
  const unsigned m0 = (unsigned)blockIdx.x * (16 * MT);
  const int n0 = wave * 16;
  const int half = lane >> 4;          // ISA A/B K-half lane split
  const int l16  = lane & 15;

  // A 16x32 bf16 fragment: lane holds row l16, K = kk + half*8 + {0..7}, +16
  const unsigned short* ap = A + (unsigned)(m0 + l16) * K + half * 8;
  // B 32x16 bf16 fragment: lane holds col l16, K = kk + half*8 + {0..7}, +16
  const unsigned short* bp = Bt + (unsigned)(n0 + l16) * K + half * 8;

  v8f c[MT];
#pragma unroll
  for (int mt = 0; mt < MT; ++mt) c[mt] = (v8f){};

#pragma unroll
  for (int kk = 0; kk < K; kk += 32) {
    v16bf b;
    ((uint4*)&b)[0] = *(const uint4*)(bp + kk);
    ((uint4*)&b)[1] = *(const uint4*)(bp + kk + 16);
#pragma unroll
    for (int mt = 0; mt < MT; ++mt) {
      v16bf a;
      ((uint4*)&a)[0] = *(const uint4*)(ap + (unsigned)mt * 16 * K + kk);
      ((uint4*)&a)[1] = *(const uint4*)(ap + (unsigned)mt * 16 * K + kk + 16);
      c[mt] = __builtin_amdgcn_wmma_f32_16x16x32_bf16(false, a, false, b,
                                                      (short)0, c[mt], false, false);
    }
  }
  // C/D layout: VGPR j -> row tileRow + j + 8*half, col n0 + l16
#pragma unroll
  for (int mt = 0; mt < MT; ++mt) {
    unsigned o = (m0 + mt * 16 + half * 8) * (unsigned)Hcols + n0 + l16;
#pragma unroll
    for (int j = 0; j < 8; ++j) out[o + (unsigned)j * Hcols] = c[mt][j];
  }
}

// Guarded single-tile variant for ragged tails (rows [0, Nrows), Nrows < 32).
template <int K>
__global__ void gemm_bf16_wmma_tail(const unsigned short* __restrict__ A,
                                    const unsigned short* __restrict__ Bt,
                                    float* __restrict__ out, int Nrows, int Hcols) {
  const int lane = threadIdx.x & 31;
  const int wave = threadIdx.x >> 5;
  const unsigned m0 = (unsigned)blockIdx.x * 16;
  const int n0 = wave * 16;
  const int half = lane >> 4;
  const int l16  = lane & 15;
  unsigned arow = m0 + l16;
  if (arow >= (unsigned)Nrows) arow = Nrows - 1;
  const unsigned short* ap = A + arow * (unsigned)K + half * 8;
  const unsigned short* bp = Bt + (unsigned)(n0 + l16) * K + half * 8;
  v8f c = {};
#pragma unroll
  for (int kk = 0; kk < K; kk += 32) {
    v16bf a, b;
    ((uint4*)&a)[0] = *(const uint4*)(ap + kk);
    ((uint4*)&a)[1] = *(const uint4*)(ap + kk + 16);
    ((uint4*)&b)[0] = *(const uint4*)(bp + kk);
    ((uint4*)&b)[1] = *(const uint4*)(bp + kk + 16);
    c = __builtin_amdgcn_wmma_f32_16x16x32_bf16(false, a, false, b,
                                                (short)0, c, false, false);
  }
#pragma unroll
  for (int j = 0; j < 8; ++j) {
    unsigned r = m0 + half * 8 + j;
    if (r < (unsigned)Nrows)
      out[r * (unsigned)Hcols + n0 + l16] = c[j];
  }
}

// ---------------------------------------------------------------- edge scatter
// One wave per edge: lanes carry one 4-float chunk each. CHUNKS_LOG2 = log2(H/4).
template <int CHUNKS_LOG2>
__global__ void edge_aggregate(const float* __restrict__ xw,
                               const int* __restrict__ src, const int* __restrict__ dst,
                               const float* __restrict__ dinv, float* __restrict__ agg,
                               long long total, int H) {
  long long idx = (long long)blockIdx.x * blockDim.x + threadIdx.x;
  if (idx >= total) return;
  long long e = idx >> CHUNKS_LOG2;
  int ch = (int)idx & ((1 << CHUNKS_LOG2) - 1);
  int s = src[e], d = dst[e];
  float coef = dinv[s] * dinv[d];
  float4 v = *(const float4*)(xw + (long long)s * H + ch * 4);
  float* o = agg + (long long)d * H + ch * 4;
  atomicAdd(o + 0, v.x * coef);
  atomicAdd(o + 1, v.y * coef);
  atomicAdd(o + 2, v.z * coef);
  atomicAdd(o + 3, v.w * coef);
}

// ---------------------------------------------------------------- fused finishers
// h = ELU(LN(agg + dinv^2*xw + bias)); write bf16 row for next GEMM.
__global__ void finish_ln_elu(const float* __restrict__ agg, const float* __restrict__ xw,
                              const float* __restrict__ dinv, const float* __restrict__ bias,
                              const float* __restrict__ lnw, const float* __restrict__ lnb,
                              unsigned short* __restrict__ hb, int N, int H) {
  int lane = threadIdx.x & 31, wv = threadIdx.x >> 5;
  int row = blockIdx.x * 8 + wv;
  if (row >= N) return;
  int c0 = lane * 4;
  unsigned base = (unsigned)row * H + c0;
  float di = dinv[row], sl = di * di;
  float4 a = *(const float4*)(agg + base);
  float4 x = *(const float4*)(xw + base);
  float v0 = a.x + sl * x.x + bias[c0 + 0], v1 = a.y + sl * x.y + bias[c0 + 1];
  float v2 = a.z + sl * x.z + bias[c0 + 2], v3 = a.w + sl * x.w + bias[c0 + 3];
  float invH = 1.0f / (float)H;
  float mu = wave_sum32(v0 + v1 + v2 + v3) * invH;
  float d0 = v0 - mu, d1 = v1 - mu, d2 = v2 - mu, d3 = v3 - mu;
  float var = wave_sum32(d0 * d0 + d1 * d1 + d2 * d2 + d3 * d3) * invH;
  float rs = rsqrtf(var + LN_EPS);
  float y0 = d0 * rs * lnw[c0 + 0] + lnb[c0 + 0];
  float y1 = d1 * rs * lnw[c0 + 1] + lnb[c0 + 1];
  float y2 = d2 * rs * lnw[c0 + 2] + lnb[c0 + 2];
  float y3 = d3 * rs * lnw[c0 + 3] + lnb[c0 + 3];
  ushort4 o;
  o.x = f2bf(y0 > 0.0f ? y0 : expm1f(y0));
  o.y = f2bf(y1 > 0.0f ? y1 : expm1f(y1));
  o.z = f2bf(y2 > 0.0f ? y2 : expm1f(y2));
  o.w = f2bf(y3 > 0.0f ? y3 : expm1f(y3));
  *(ushort4*)(hb + base) = o;
}

// Layer-2 finisher: LN+ELU, then h_b = RR*h + (1-RR)*bcos(h), pool into sums/cnts.
// With B == 1.0 the bcos power term (|xs|+eps)^0 == 1 exactly, so bcos(h) = TEMP*h/(nrm+eps).
__global__ void finish2_pool(const float* __restrict__ agg, const float* __restrict__ xw,
                             const float* __restrict__ dinv, const float* __restrict__ bias,
                             const float* __restrict__ lnw, const float* __restrict__ lnb,
                             const int* __restrict__ batch, float* __restrict__ sums,
                             float* __restrict__ cnts, int N, int H) {
  int lane = threadIdx.x & 31, wv = threadIdx.x >> 5;
  int row = blockIdx.x * 8 + wv;
  if (row >= N) return;
  int c0 = lane * 4;
  unsigned base = (unsigned)row * H + c0;
  float di = dinv[row], sl = di * di;
  float4 a = *(const float4*)(agg + base);
  float4 x = *(const float4*)(xw + base);
  float v0 = a.x + sl * x.x + bias[c0 + 0], v1 = a.y + sl * x.y + bias[c0 + 1];
  float v2 = a.z + sl * x.z + bias[c0 + 2], v3 = a.w + sl * x.w + bias[c0 + 3];
  float invH = 1.0f / (float)H;
  float mu = wave_sum32(v0 + v1 + v2 + v3) * invH;
  float d0 = v0 - mu, d1 = v1 - mu, d2 = v2 - mu, d3 = v3 - mu;
  float var = wave_sum32(d0 * d0 + d1 * d1 + d2 * d2 + d3 * d3) * invH;
  float rs = rsqrtf(var + LN_EPS);
  float h[4];
  float dd[4] = { d0, d1, d2, d3 };
#pragma unroll
  for (int j = 0; j < 4; ++j) {
    float y = dd[j] * rs * lnw[c0 + j] + lnb[c0 + j];
    h[j] = y > 0.0f ? y : expm1f(y);
  }
  float nrm = sqrtf(wave_sum32(h[0] * h[0] + h[1] * h[1] + h[2] * h[2] + h[3] * h[3]) + BCOS_EPS);
  float s = TEMP_C / (nrm + BCOS_EPS);
  int b = batch[row];
  float* o = sums + (unsigned)b * H + c0;
#pragma unroll
  for (int j = 0; j < 4; ++j) {
    float hb = RR_C * h[j] + (1.0f - RR_C) * (s * h[j]);   // bcos == s*h for B=1
    atomicAdd(o + j, hb);
  }
  if (lane == 0) atomicAdd(cnts + b, 1.0f);
}

// ---------------------------------------------------------------- classifier
__global__ void prep_wn(const float* __restrict__ cls_v, const float* __restrict__ cls_g,
                        float* __restrict__ Wn, int C, int H) {
  int lane = threadIdx.x & 31, c = threadIdx.x >> 5;
  if (c >= C) return;
  int c0 = lane * 4;
  float4 v = *(const float4*)(cls_v + c * H + c0);
  float n2 = wave_sum32(v.x * v.x + v.y * v.y + v.z * v.z + v.w * v.w);
  float s = cls_g[c] * rsqrtf(n2);
  float4 o = { v.x * s, v.y * s, v.z * s, v.w * s };
  *(float4*)(Wn + c * H + c0) = o;
}

__global__ void classify(const float* __restrict__ sums, const float* __restrict__ cnts,
                         const float* __restrict__ Wn, const float* __restrict__ cls_b,
                         float* __restrict__ out, int C, int H) {
  int g = blockIdx.x, lane = threadIdx.x;
  int c0 = lane * 4;
  float inv = 1.0f / fmaxf(cnts[g], 1.0f);
  float4 v = *(const float4*)(sums + (unsigned)g * H + c0);
  float g0 = v.x * inv, g1 = v.y * inv, g2 = v.z * inv, g3 = v.w * inv;
  for (int c = 0; c < C; ++c) {
    const float* w = Wn + c * H + c0;
    float p = wave_sum32(g0 * w[0] + g1 * w[1] + g2 * w[2] + g3 * w[3]);
    if (lane == 0) out[g * C + c] = p + cls_b[c];
  }
}

// ---------------------------------------------------------------- launcher
static inline int grd(long long n, int t) { return (int)((n + t - 1) / t); }

extern "C" void kernel_launch(void* const* d_in, const int* in_sizes, int n_in,
                              void* d_out, int out_size, void* d_ws, size_t ws_size,
                              hipStream_t stream) {
  const float* x     = (const float*)d_in[0];
  const int*   src   = (const int*)d_in[1];
  const int*   dst   = (const int*)d_in[2];
  const int*   batch = (const int*)d_in[3];
  const float* W1    = (const float*)d_in[4];
  const float* b1    = (const float*)d_in[5];
  const float* ln1w  = (const float*)d_in[6];
  const float* ln1b  = (const float*)d_in[7];
  const float* W2    = (const float*)d_in[8];
  const float* b2    = (const float*)d_in[9];
  const float* ln2w  = (const float*)d_in[10];
  const float* ln2b  = (const float*)d_in[11];
  const float* clsv  = (const float*)d_in[12];
  const float* clsg  = (const float*)d_in[13];
  const float* clsb  = (const float*)d_in[14];
  float* out = (float*)d_out;

  const int N = in_sizes[3];
  const int E = in_sizes[1];
  const int H = in_sizes[5];            // 128
  const int F = in_sizes[0] / N;        // 128
  const int C = in_sizes[13];           // 10
  const int G = out_size / C;           // 512

  // workspace carve-out (256B aligned)
  size_t off = 0;
  auto carve = [&](size_t bytes) { size_t o = off; off += (bytes + 255) & ~(size_t)255; return o; };
  char* ws = (char*)d_ws;
  float*          dinv = (float*)(ws + carve((size_t)N * 4));
  unsigned short* xb   = (unsigned short*)(ws + carve((size_t)N * F * 2)); // x/h bf16
  float*          xw   = (float*)(ws + carve((size_t)N * H * 4));
  float*          agg  = (float*)(ws + carve((size_t)N * H * 4));
  float*          sums = (float*)(ws + carve((size_t)G * H * 4));
  float*          cnts = (float*)(ws + carve((size_t)G * 4));
  unsigned short* Wt1  = (unsigned short*)(ws + carve((size_t)H * F * 2));
  unsigned short* Wt2  = (unsigned short*)(ws + carve((size_t)H * H * 2));
  float*          Wn   = (float*)(ws + carve((size_t)C * H * 4));
  (void)ws_size; (void)n_in;

  const long long NH = (long long)N * H;
  const long long NF = (long long)N * F;
  const int Nmain = N & ~31;           // whole 32-row blocks (N=100000 -> all of it)
  const int Nrem  = N - Nmain;
  const long long Etot = (long long)E * (H >> 2);

  // degrees -> dinv  (N assumed %4==0 for the vec4 zero; true here)
  zero_f32v4<<<grd(N / 4, 256), 256, 0, stream>>>((float4*)dinv, N / 4);
  edge_degree<<<grd(E, 256), 256, 0, stream>>>(dst, dinv, E);
  degree_to_dinv<<<grd(N, 256), 256, 0, stream>>>(dinv, N);

  // operand prep
  cvt_f32_to_bf16v4<<<grd(NF / 4, 256), 256, 0, stream>>>((const float4*)x, (ushort4*)xb, NF / 4);
  transpose_w_bf16<<<grd(F * H, 256), 256, 0, stream>>>(W1, Wt1, F, H);
  transpose_w_bf16<<<grd(H * H, 256), 256, 0, stream>>>(W2, Wt2, H, H);

  // ---- layer 1
  zero_f32v4<<<grd(NH / 4, 256), 256, 0, stream>>>((float4*)agg, NH / 4);
  if (Nmain > 0)
    gemm_bf16_wmma<128, 2><<<Nmain / 32, 256, 0, stream>>>(xb, Wt1, xw, H);
  if (Nrem > 0)
    gemm_bf16_wmma_tail<128><<<grd(Nrem, 16), 256, 0, stream>>>(
        xb + (size_t)Nmain * F, Wt1, xw + (size_t)Nmain * H, Nrem, H);
  edge_aggregate<5><<<grd(Etot, 256), 256, 0, stream>>>(xw, src, dst, dinv, agg, Etot, H);
  finish_ln_elu<<<grd(N, 8), 256, 0, stream>>>(agg, xw, dinv, b1, ln1w, ln1b, xb, N, H);

  // ---- layer 2
  zero_f32v4<<<grd(NH / 4, 256), 256, 0, stream>>>((float4*)agg, NH / 4);
  if (Nmain > 0)
    gemm_bf16_wmma<128, 2><<<Nmain / 32, 256, 0, stream>>>(xb, Wt2, xw, H);
  if (Nrem > 0)
    gemm_bf16_wmma_tail<128><<<grd(Nrem, 16), 256, 0, stream>>>(
        xb + (size_t)Nmain * H, Wt2, xw + (size_t)Nmain * H, Nrem, H);
  edge_aggregate<5><<<grd(Etot, 256), 256, 0, stream>>>(xw, src, dst, dinv, agg, Etot, H);

  // ---- bcos blend + pool
  zero_f32v4<<<grd((long long)G * H / 4, 256), 256, 0, stream>>>((float4*)sums, (long long)G * H / 4);
  zero_f32v4<<<grd(G / 4, 256), 256, 0, stream>>>((float4*)cnts, G / 4);
  finish2_pool<<<grd(N, 8), 256, 0, stream>>>(agg, xw, dinv, b2, ln2w, ln2b, batch, sums, cnts, N, H);

  // ---- weight-normalized classifier
  prep_wn<<<1, ((C * 32 + 31) / 32) * 32, 0, stream>>>(clsv, clsg, Wn, C, H);
  classify<<<G, 32, 0, stream>>>(sums, cnts, Wn, clsb, out, C, H);
}
